// TemporalGNN_36060545417856
// MI455X (gfx1250) — compile-verified, matrix operand order checked by reading, not run on hardware
//
#include <hip/hip_runtime.h>

typedef __attribute__((ext_vector_type(16))) __bf16 v16bf;
typedef __attribute__((ext_vector_type(8)))  float  v8f;

#define B_    4
#define T_    16
#define N_    4096
#define NF_   8
#define H_    64
#define C_    13
#define E_    131072
#define M_    (B_*N_)        // 16384 sequences
#define ROWS_ (B_*T_*N_)     // 262144 encoder rows
#define G4_   (4*H_)         // 256 gate outputs

// ---- workspace layout (bytes) ----
#define OFF_H1    ((size_t)0)                               // 67 MB  (enc hidden1, later LSTM-L0 out)
#define OFF_HENC  (OFF_H1   + (size_t)ROWS_*H_*4)           // 67 MB  (enc out, later LSTM-L1 out)
#define OFF_XSEQ  (OFF_HENC + (size_t)ROWS_*H_*4)           // 67 MB  (LSTM-L0 input, (M,T,H))
#define OFF_ACCUM (OFF_XSEQ + (size_t)M_*T_*H_*4)           // 4 MB   (scatter accumulator (B,N,H))
#define OFF_CNT   (OFF_ACCUM+ (size_t)B_*N_*H_*4)           // 16 KB  (per-dst counts)
#define OFF_WBF   (OFF_CNT  + (size_t)N_*4)                 // bf16 weights

// -------------------- helpers --------------------
__device__ __host__ inline __bf16 f2bf(float f) {
  union { float f; unsigned u; } v; v.f = f;
  unsigned r = v.u + 0x7FFFu + ((v.u >> 16) & 1u);   // round-to-nearest-even
  unsigned short h = (unsigned short)(r >> 16);
  return __builtin_bit_cast(__bf16, h);
}

__device__ inline float sigmf(float x) { return 1.0f / (1.0f + __expf(-x)); }

__device__ inline v8f wmma_bf16(v16bf a, v16bf b, v8f c) {
  return __builtin_amdgcn_wmma_f32_16x16x32_bf16(
      /*neg_a=*/false, a, /*neg_b=*/false, b,
      /*c_mod=*/(short)0, c, /*reuse_a=*/false, /*reuse_b=*/false);
}

// A fragment (16x32 bf16, MxK) from a 64-wide bf16 row in LDS.
// Lane layout (ISA 7.12.2): lanes 0-15 -> K = k0+{0..7, 16..23};
// lanes 16-31 -> K = k0+{8..15, 24..31}. Contiguous 8-element (16 B) runs.
__device__ inline v16bf make_afrag_bf(const __bf16* base, int k0, int hi) {
  v16bf a;
  const __bf16* p = base + k0 + hi * 8;
#pragma unroll
  for (int e = 0; e < 8; ++e) { a[e] = p[e]; a[e + 8] = p[e + 16]; }
  return a;
}

// B fragment (32x16 bf16, KxN) from W[nOut][K] row-major bf16 (B = W^T),
// lane = output column; same K striping as A.
__device__ inline v16bf make_bfrag(const __bf16* w, int ocol, int K, int k0, int hi) {
  v16bf b;
  const __bf16* p = w + (size_t)ocol * K + k0 + hi * 8;
#pragma unroll
  for (int e = 0; e < 8; ++e) { b[e] = p[e]; b[e + 8] = p[e + 16]; }
  return b;
}

// -------------------- kernels --------------------
__global__ void cvt_bf16_kernel(const float* __restrict__ in, __bf16* __restrict__ out, int n) {
  int i = blockIdx.x * 256 + threadIdx.x;
  if (i < n) out[i] = f2bf(in[i]);
}

// encoder layer 1: h1 = relu(nf @ sc_w1^T + b1), K = 8 (VALU is fine)
__global__ __launch_bounds__(256) void enc1_kernel(
    const float* __restrict__ nf, const float* __restrict__ w1,
    const float* __restrict__ b1, float* __restrict__ h1) {
  int idx = blockIdx.x * 256 + threadIdx.x;       // over ROWS_*64
  int row = idx >> 6, j = idx & 63;
  const float* x = nf + (size_t)row * NF_;
  const float* w = w1 + (size_t)j * NF_;
  float acc = b1[j];
#pragma unroll
  for (int k = 0; k < NF_; ++k) acc += x[k] * w[k];
  h1[(size_t)row * H_ + j] = fmaxf(acc, 0.0f);
}

// generic (rows,64) @ (64,64)^T + bias via bf16 WMMA (encoder layer 2).
// A tile staged in LDS as bf16 (one conversion per element).
__global__ __launch_bounds__(128) void gemm64_kernel(
    const float* __restrict__ A, const __bf16* __restrict__ W,
    const float* __restrict__ bias, float* __restrict__ out, int rows) {
  __shared__ __bf16 At[16][H_];
  int tid = threadIdx.x, lane = tid & 31, wave = tid >> 5;
  int row0 = blockIdx.x * 16;
  if (row0 >= rows) return;
  for (int i = tid; i < 16 * H_; i += 128) {
    int rr = i >> 6, cc = i & 63;
    At[rr][cc] = f2bf(A[(size_t)(row0 + rr) * H_ + cc]);
  }
  __syncthreads();
  int r = lane & 15, hi = lane >> 4;
  int n0 = wave * 16;
  v8f acc = {};
#pragma unroll
  for (int k0 = 0; k0 < H_; k0 += 32) {
    v16bf a = make_afrag_bf(&At[r][0], k0, hi);
    v16bf b = make_bfrag(W, n0 + r, H_, k0, hi);
    acc = wmma_bf16(a, b, acc);
  }
  int col = n0 + r;
  float bv = bias[col];
#pragma unroll
  for (int v = 0; v < 8; ++v) {
    int rr = v + hi * 8;
    out[(size_t)(row0 + rr) * H_ + col] = acc[v] + bv;
  }
}

// message passing: scatter-add per edge (one timestep)
__global__ __launch_bounds__(256) void scatter_kernel(
    const float* __restrict__ henc, const int* __restrict__ ei,
    float* __restrict__ accum, float* __restrict__ cnt, int t) {
  int idx = blockIdx.x * 256 + threadIdx.x;       // over E_*B_
  int e = idx >> 2, b = idx & 3;
  int src = ei[(size_t)t * 2 * E_ + e] & (N_ - 1);
  int dst = ei[(size_t)t * 2 * E_ + E_ + e] & (N_ - 1);
  const float* s = henc + (((size_t)b * T_ + t) * N_ + src) * H_;
  float* d = accum + ((size_t)b * N_ + dst) * H_;
#pragma unroll 8
  for (int h = 0; h < H_; ++h) atomicAdd(&d[h], s[h]);
  if (b == 0) atomicAdd(&cnt[dst], 1.0f);
}

// message passing: combine + reorder to (M, T, H)
__global__ __launch_bounds__(256) void combine_kernel(
    const float* __restrict__ henc, const float* __restrict__ accum,
    const float* __restrict__ cnt, float* __restrict__ xseq, int t) {
  int idx = blockIdx.x * 256 + threadIdx.x;       // over B_*N_*H_
  int h = idx & 63, n = (idx >> 6) & (N_ - 1), b = idx >> 18;
  float c = cnt[n];
  float hv = henc[(((size_t)b * T_ + t) * N_ + n) * H_ + h];
  float mean = accum[((size_t)b * N_ + n) * H_ + h] / fmaxf(c, 1.0f);
  float res = (c > 0.0f) ? (hv + mean) * 0.5f : hv;
  xseq[(((size_t)b * N_ + n) * T_ + t) * H_ + h] = res;
}

// fused 1-layer LSTM over 16 sequences/block; both projections via WMMA.
// x_t and h_{t-1} staged in LDS as bf16; A fragments hoisted and reused
// across all 4 gate-column tiles per wave.
__global__ __launch_bounds__(128) void lstm_kernel(
    const float* __restrict__ xin,                 // (M, T, 64)
    const __bf16* __restrict__ wih,                // (256, 64)
    const __bf16* __restrict__ whh,                // (256, 64)
    const float* __restrict__ bih, const float* __restrict__ bhh,
    float* __restrict__ hout) {                    // (M, T, 64)
  __shared__ __bf16 xtb[16][H_];
  __shared__ __bf16 hb[16][H_];
  __shared__ float  cst[16][H_];
  __shared__ float  gates[16][G4_];
  __shared__ float  bsum[G4_];

  int tid = threadIdx.x, lane = tid & 31, wave = tid >> 5;
  int m0 = blockIdx.x * 16;
  int r = lane & 15, hi = lane >> 4;
  const __bf16 bz = f2bf(0.0f);

  for (int i = tid; i < G4_; i += 128) bsum[i] = bih[i] + bhh[i];
  for (int i = tid; i < 16 * H_; i += 128) {
    (&hb[0][0])[i] = bz;
    (&cst[0][0])[i] = 0.0f;
  }
  __syncthreads();

  for (int t = 0; t < T_; ++t) {
    // stage x_t tile as bf16 (one conversion per element)
    for (int i = tid; i < 16 * H_; i += 128) {
      int mm = i >> 6, h = i & 63;
      xtb[mm][h] = f2bf(xin[(((size_t)(m0 + mm)) * T_ + t) * H_ + h]);
    }
    // hint the next timestep's tile into cache while we compute
    if (t + 1 < T_ && tid < 16)
      __builtin_prefetch(xin + (((size_t)(m0 + tid)) * T_ + (t + 1)) * H_, 0, 1);
    __syncthreads();

    // A fragments are shared by all 4 column tiles of this wave
    v16bf ax0 = make_afrag_bf(&xtb[r][0], 0,  hi);
    v16bf ax1 = make_afrag_bf(&xtb[r][0], 32, hi);
    v16bf ah0 = make_afrag_bf(&hb[r][0],  0,  hi);
    v16bf ah1 = make_afrag_bf(&hb[r][0],  32, hi);

#pragma unroll
    for (int q = 0; q < 4; ++q) {
      int n0 = (wave * 4 + q) * 16;
      int col = n0 + r;
      v8f acc = {};
      acc = wmma_bf16(ax0, make_bfrag(wih, col, H_, 0,  hi), acc);
      acc = wmma_bf16(ax1, make_bfrag(wih, col, H_, 32, hi), acc);
      acc = wmma_bf16(ah0, make_bfrag(whh, col, H_, 0,  hi), acc);
      acc = wmma_bf16(ah1, make_bfrag(whh, col, H_, 32, hi), acc);
      float bv = bsum[col];
#pragma unroll
      for (int v = 0; v < 8; ++v) gates[v + hi * 8][col] = acc[v] + bv;
    }
    __syncthreads();

    for (int i = tid; i < 16 * H_; i += 128) {
      int mm = i >> 6, h = i & 63;
      float ig = sigmf(gates[mm][h]);
      float fg = sigmf(gates[mm][H_ + h]);
      float gg = tanhf(gates[mm][2 * H_ + h]);
      float og = sigmf(gates[mm][3 * H_ + h]);
      float c  = fg * cst[mm][h] + ig * gg;
      float hn = og * tanhf(c);
      cst[mm][h] = c;
      hb[mm][h] = f2bf(hn);
      hout[(((size_t)(m0 + mm)) * T_ + t) * H_ + h] = hn;
    }
    __syncthreads();
  }
}

// fused attention (last-query only) + out-proj + 3-layer MLP head
__global__ __launch_bounds__(128) void attn_mlp_kernel(
    const float* __restrict__ xseq,                // (M, 16, 64)
    const float* __restrict__ aw, const float* __restrict__ ab,
    const float* __restrict__ ow, const float* __restrict__ ob,
    const float* __restrict__ pw1, const float* __restrict__ pb1,
    const float* __restrict__ pw2, const float* __restrict__ pb2,
    const float* __restrict__ pw3, const float* __restrict__ pb3,
    float* __restrict__ out) {                     // (M, 13)
  __shared__ float X[2][16][H_];
  __shared__ float Kl[2][16][H_];
  __shared__ float Vl[2][16][H_];
  __shared__ float q15[2][H_];
  __shared__ float sc[2][4][16];
  __shared__ float pr[2][4][16];
  __shared__ float o15[2][H_];
  __shared__ float fh[2][H_];
  __shared__ float z1[2][2 * H_];
  __shared__ float z2[2][H_];

  int tid = threadIdx.x;
  int s = tid >> 6, j = tid & 63;
  size_t m = (size_t)blockIdx.x * 2 + s;

  for (int i = j; i < 16 * H_; i += 64)
    (&X[s][0][0])[i] = xseq[m * 16 * H_ + i];
  __syncthreads();

  // K, V for all t; Q only at t = 15
  for (int t = 0; t < 16; ++t) {
    float accK = ab[H_ + j], accV = ab[2 * H_ + j];
#pragma unroll 8
    for (int k = 0; k < H_; ++k) {
      float xv = X[s][t][k];
      accK += xv * aw[(size_t)(H_ + j) * H_ + k];
      accV += xv * aw[(size_t)(2 * H_ + j) * H_ + k];
    }
    Kl[s][t][j] = accK;
    Vl[s][t][j] = accV;
  }
  {
    float accQ = ab[j];
#pragma unroll 8
    for (int k = 0; k < H_; ++k) accQ += X[s][15][k] * aw[(size_t)j * H_ + k];
    q15[s][j] = accQ;
  }
  __syncthreads();

  int head = j >> 4, tt = j & 15;
  {
    float acc = 0.0f;
#pragma unroll
    for (int d = 0; d < 16; ++d) acc += q15[s][head * 16 + d] * Kl[s][tt][head * 16 + d];
    sc[s][head][tt] = acc * 0.25f;                 // 1/sqrt(16)
  }
  __syncthreads();
  {
    float mx = -1e30f;
#pragma unroll
    for (int t2 = 0; t2 < 16; ++t2) mx = fmaxf(mx, sc[s][head][t2]);
    float den = 0.0f;
#pragma unroll
    for (int t2 = 0; t2 < 16; ++t2) den += __expf(sc[s][head][t2] - mx);
    pr[s][head][tt] = __expf(sc[s][head][tt] - mx) / den;
  }
  __syncthreads();
  {
    float acc = 0.0f;
#pragma unroll
    for (int t2 = 0; t2 < 16; ++t2) acc += pr[s][head][t2] * Vl[s][t2][j];
    o15[s][j] = acc;
  }
  __syncthreads();
  {
    float acc = ob[j];
#pragma unroll 8
    for (int k = 0; k < H_; ++k) acc += o15[s][k] * ow[(size_t)j * H_ + k];
    fh[s][j] = acc;
  }
  __syncthreads();
  for (int rr = j; rr < 2 * H_; rr += 64) {
    float acc = pb1[rr];
#pragma unroll 8
    for (int k = 0; k < H_; ++k) acc += fh[s][k] * pw1[(size_t)rr * H_ + k];
    z1[s][rr] = fmaxf(acc, 0.0f);
  }
  __syncthreads();
  {
    float acc = pb2[j];
#pragma unroll 8
    for (int k = 0; k < 2 * H_; ++k) acc += z1[s][k] * pw2[(size_t)j * 2 * H_ + k];
    z2[s][j] = fmaxf(acc, 0.0f);
  }
  __syncthreads();
  if (j < C_) {
    float acc = pb3[j];
#pragma unroll 8
    for (int k = 0; k < H_; ++k) acc += z2[s][k] * pw3[(size_t)j * H_ + k];
    out[m * C_ + j] = acc;
  }
}

// -------------------- host --------------------
extern "C" void kernel_launch(void* const* d_in, const int* in_sizes, int n_in,
                              void* d_out, int out_size, void* d_ws, size_t ws_size,
                              hipStream_t stream) {
  const float* nf   = (const float*)d_in[0];
  const int*   ei   = (const int*)d_in[1];
  const float* scw1 = (const float*)d_in[2];
  const float* scb1 = (const float*)d_in[3];
  const float* scw2 = (const float*)d_in[4];
  const float* scb2 = (const float*)d_in[5];
  const float* wih0 = (const float*)d_in[6];
  const float* whh0 = (const float*)d_in[7];
  const float* bih0 = (const float*)d_in[8];
  const float* bhh0 = (const float*)d_in[9];
  const float* wih1 = (const float*)d_in[10];
  const float* whh1 = (const float*)d_in[11];
  const float* bih1 = (const float*)d_in[12];
  const float* bhh1 = (const float*)d_in[13];
  const float* aw   = (const float*)d_in[14];
  const float* ab   = (const float*)d_in[15];
  const float* ow   = (const float*)d_in[16];
  const float* ob   = (const float*)d_in[17];
  const float* pw1  = (const float*)d_in[18];
  const float* pb1  = (const float*)d_in[19];
  const float* pw2  = (const float*)d_in[20];
  const float* pb2  = (const float*)d_in[21];
  const float* pw3  = (const float*)d_in[22];
  const float* pb3  = (const float*)d_in[23];

  char* ws = (char*)d_ws;
  float*  h1    = (float*)(ws + OFF_H1);     // also LSTM-L0 output
  float*  henc  = (float*)(ws + OFF_HENC);   // also LSTM-L1 output
  float*  xseq  = (float*)(ws + OFF_XSEQ);
  float*  accum = (float*)(ws + OFF_ACCUM);
  float*  cnt   = (float*)(ws + OFF_CNT);
  __bf16* wih0b = (__bf16*)(ws + OFF_WBF);
  __bf16* whh0b = wih0b + G4_ * H_;
  __bf16* wih1b = whh0b + G4_ * H_;
  __bf16* whh1b = wih1b + G4_ * H_;
  __bf16* scw2b = whh1b + G4_ * H_;

  // 0) convert GEMM weights to bf16 once
  cvt_bf16_kernel<<<(G4_ * H_ + 255) / 256, 256, 0, stream>>>(wih0, wih0b, G4_ * H_);
  cvt_bf16_kernel<<<(G4_ * H_ + 255) / 256, 256, 0, stream>>>(whh0, whh0b, G4_ * H_);
  cvt_bf16_kernel<<<(G4_ * H_ + 255) / 256, 256, 0, stream>>>(wih1, wih1b, G4_ * H_);
  cvt_bf16_kernel<<<(G4_ * H_ + 255) / 256, 256, 0, stream>>>(whh1, whh1b, G4_ * H_);
  cvt_bf16_kernel<<<(H_ * H_ + 255) / 256, 256, 0, stream>>>(scw2, scw2b, H_ * H_);

  // 1) encoder
  enc1_kernel<<<(ROWS_ * H_) / 256, 256, 0, stream>>>(nf, scw1, scb1, h1);
  gemm64_kernel<<<ROWS_ / 16, 128, 0, stream>>>(h1, scw2b, scb2, henc, ROWS_);

  // 2) message passing, one independent pass per timestep (reuses 4 MB accumulator)
  for (int t = 0; t < T_; ++t) {
    hipMemsetAsync(ws + OFF_ACCUM, 0, (size_t)B_ * N_ * H_ * 4 + (size_t)N_ * 4, stream);
    scatter_kernel<<<(E_ * B_) / 256, 256, 0, stream>>>(henc, ei, accum, cnt, t);
    combine_kernel<<<(B_ * N_ * H_) / 256, 256, 0, stream>>>(henc, accum, cnt, xseq, t);
  }

  // 3) two fused LSTM layers (WMMA for both x and h projections)
  lstm_kernel<<<M_ / 16, 128, 0, stream>>>(xseq, wih0b, whh0b, bih0, bhh0, h1);
  lstm_kernel<<<M_ / 16, 128, 0, stream>>>(h1,   wih1b, whh1b, bih1, bhh1, henc);

  // 4) attention (last query only) + output projection + MLP head
  attn_mlp_kernel<<<M_ / 2, 128, 0, stream>>>(henc, aw, ab, ow, ob,
                                              pw1, pb1, pw2, pb2, pw3, pb3,
                                              (float*)d_out);
}
